// FrameMemoryRetrieval_56023553409095
// MI455X (gfx1250) — matrix operand order repaired and checked by reading, not compiled, not run on hardware
//
#include <hip/hip_runtime.h>

// ---------------------------------------------------------------------------
// Shapes (compile-time constants from the reference)
//   B=1, I=256, J=64, N=384, C=256, H=8, CH=32
//   rows_q = B*I*N = 98304, rows_kv = J*N = 24576
// ---------------------------------------------------------------------------
#define NSEQ   384
#define CDIM   256
#define ROWS_Q 98304
#define ROWS_KV 24576

typedef __attribute__((ext_vector_type(16))) __bf16 v16bf;
typedef __attribute__((ext_vector_type(8)))  __bf16 v8bf;
typedef __attribute__((ext_vector_type(8)))  float  v8f;
typedef __attribute__((ext_vector_type(4)))  int    v4i;

#ifndef __has_builtin
#define __has_builtin(x) 0
#endif
#if __has_builtin(__builtin_amdgcn_global_load_async_to_lds_b128)
#define HAVE_ASYNC_LDS 1
#else
#define HAVE_ASYNC_LDS 0
#endif

union V16U { v16bf v; v8bf h[2]; };

__device__ static inline v8f vz8() {
    v8f z;
#pragma unroll
    for (int i = 0; i < 8; ++i) z[i] = 0.0f;
    return z;
}

// ---------------------------------------------------------------------------
// Weight transpose + bf16 convert:  WT[n*K + k] = bf16(W[k*N + n])
// ---------------------------------------------------------------------------
__global__ void wtrans_kernel(const float* __restrict__ W, __bf16* __restrict__ WT,
                              int K, int N) {
    int idx = blockIdx.x * 256 + threadIdx.x;
    if (idx >= K * N) return;
    int n = idx / K, k = idx % K;
    WT[idx] = (__bf16)W[(size_t)k * N + n];
}

// ---------------------------------------------------------------------------
// LayerNorm over C=256, one wave (32 lanes) per row, bf16 output.
// ---------------------------------------------------------------------------
__global__ __launch_bounds__(256)
void ln256_kernel(const float* __restrict__ x, const float* __restrict__ scale,
                  const float* __restrict__ bias, __bf16* __restrict__ out, int rows) {
    const int lane = threadIdx.x & 31;
    const int row  = blockIdx.x * 8 + (threadIdx.x >> 5);
    if (row >= rows) return;
    const float* xr = x + (size_t)row * CDIM;
    float v[8];
    float s = 0.0f;
#pragma unroll
    for (int t = 0; t < 8; ++t) { v[t] = xr[lane + 32 * t]; s += v[t]; }
#pragma unroll
    for (int off = 16; off > 0; off >>= 1) s += __shfl_xor(s, off, 32);
    const float mu = s * (1.0f / CDIM);
    float vs = 0.0f;
#pragma unroll
    for (int t = 0; t < 8; ++t) { float d = v[t] - mu; vs += d * d; }
#pragma unroll
    for (int off = 16; off > 0; off >>= 1) vs += __shfl_xor(vs, off, 32);
    const float inv = rsqrtf(vs * (1.0f / CDIM) + 1e-5f);
    __bf16* orow = out + (size_t)row * CDIM;
#pragma unroll
    for (int t = 0; t < 8; ++t) {
        int c = lane + 32 * t;
        orow[c] = (__bf16)((v[t] - mu) * inv * scale[c] + bias[c]);
    }
}

// ---------------------------------------------------------------------------
// Generic WMMA GEMM:  out(MxN) = A(MxK, bf16 row-major) x WT(NxK, bf16)^T
// Block = 128 threads (4 waves), tile M=64 N=64, each wave does 16x64.
// MODE 0: store bf16 (optional bias / relu)
// MODE 1: store f32 resid + v * rmask[row]   (masked residual epilogue)
// ---------------------------------------------------------------------------
template <bool RELU, bool BIAS, int MODE>
__global__ __launch_bounds__(128)
void gemm_bf16_kernel(const __bf16* __restrict__ A, const __bf16* __restrict__ WT,
                      const float* __restrict__ bias, const float* __restrict__ resid,
                      const int* __restrict__ rmask, float* __restrict__ outf,
                      __bf16* __restrict__ outb, int M, int N, int K) {
    const int lane = threadIdx.x & 31;
    const int wave = threadIdx.x >> 5;
    const int l15  = lane & 15;
    const int lh   = lane >> 4;           // lane half (0/1)
    const int m0   = blockIdx.x * 64 + wave * 16;
    const int n0   = blockIdx.y * 64;
    const size_t arow = (size_t)(m0 + l15) * K;

    v8f acc[4] = {vz8(), vz8(), vz8(), vz8()};

    for (int k0 = 0; k0 < K; k0 += 32) {
        __builtin_prefetch(A + arow + k0 + 256, 0, 1);   // global_prefetch_b8
        V16U a;
        a.h[0] = *(const v8bf*)(A + arow + k0 + lh * 16);
        a.h[1] = *(const v8bf*)(A + arow + k0 + 8 + lh * 16);
#pragma unroll
        for (int nf = 0; nf < 4; ++nf) {
            const v16bf b = *(const v16bf*)(WT + (size_t)(n0 + nf * 16 + l15) * K +
                                            k0 + lh * 16);
            acc[nf] = __builtin_amdgcn_wmma_f32_16x16x32_bf16(
                false, a.v, false, b, (short)0, acc[nf], false, false);
        }
    }

#pragma unroll
    for (int nf = 0; nf < 4; ++nf) {
#pragma unroll
        for (int r = 0; r < 8; ++r) {
            const int row = m0 + r + lh * 8;
            const int col = n0 + nf * 16 + l15;
            float v = acc[nf][r];
            if (BIAS) v += bias[col];
            if (RELU) v = v > 0.0f ? v : 0.0f;
            const size_t off = (size_t)row * N + col;
            if (MODE == 0) {
                outb[off] = (__bf16)v;
            } else {
                outf[off] = resid[off] + v * (float)rmask[row];
            }
        }
    }
}

// ---------------------------------------------------------------------------
// Fused attention per (n, i-tile of 64):
//   stage v rows coalesced into LDS (async global->LDS when available),
//   transpose to v^T inside LDS,
//   S[i,j,h] = q . k   (one WMMA per 16x16 tile per head, K = CH = 32),
//   softmax over h in registers (additive mask is constant over h -> cancels),
//   out[i,h,c] = sum_j S * v  via WMMA from LDS, sigmoid gating, bf16 store.
// Dynamic LDS: s_s [8][64][64] bf16 (64KB, aliased by the 32KB v staging
// buffer during the prologue) + s_vt [8][32][64] bf16 (32KB).
// ---------------------------------------------------------------------------
__global__ __launch_bounds__(256)
void attn_kernel(const __bf16* __restrict__ q, const __bf16* __restrict__ kv,
                 const __bf16* __restrict__ gate, __bf16* __restrict__ att) {
    extern __shared__ char smem[];
    __bf16* s_s  = (__bf16*)smem;                       // [h][i][j] 8*64*64
    __bf16* s_vt = (__bf16*)(smem + 8 * 64 * 64 * 2);   // [h][c][j] 8*32*64
    __bf16* s_v  = (__bf16*)smem;                       // [j][c] 64*256 (alias,
                                                        //  dead before s_s use)
    const int n     = blockIdx.y;
    const int itile = blockIdx.x;
    const int tid   = threadIdx.x;
    const int lane  = tid & 31;
    const int wave  = tid >> 5;
    const int l15   = lane & 15;
    const int lh    = lane >> 4;

    // ---- stage v rows: s_v[j][c] = v[j, n, :, :]  (16B per lane, coalesced)
#if HAVE_ASYNC_LDS
    for (int u = tid; u < 2048; u += 256) {
        const int j = u >> 5, seg = (u & 31) * 8;
        __builtin_amdgcn_global_load_async_to_lds_b128(
            (v4i*)(kv + ((size_t)(j * NSEQ + n)) * 512 + 256 + seg),
            (v4i*)(s_v + j * 256 + seg), 0, 0);
    }
#if __has_builtin(__builtin_amdgcn_s_wait_asynccnt)
    __builtin_amdgcn_s_wait_asynccnt(0);
#else
    asm volatile("s_wait_asynccnt 0" ::: "memory");
#endif
#else
    for (int u = tid; u < 2048; u += 256) {
        const int j = u >> 5, seg = (u & 31) * 8;
        *(v8bf*)(s_v + j * 256 + seg) =
            *(const v8bf*)(kv + ((size_t)(j * NSEQ + n)) * 512 + 256 + seg);
    }
#endif
    __syncthreads();

    // ---- in-LDS transpose: s_vt[h][c][j] = s_v[j][h*32+c] ----
    for (int e = tid; e < 8 * 32 * 64; e += 256) {
        const int h = e >> 11, c = (e >> 6) & 31, j = e & 63;
        s_vt[e] = s_v[j * 256 + h * 32 + c];
    }
    __syncthreads();

    // ---- scores + softmax over h (16 (i,j) tiles; 2 per wave) ----
    for (int t = wave; t < 16; t += 8) {
        const int ifr = t >> 2, jfr = t & 3;
        const int qi = itile * 64 + ifr * 16 + l15;
        const size_t qbase = ((size_t)qi * NSEQ + n) * CDIM;
        const size_t kbase = ((size_t)((jfr * 16 + l15) * NSEQ + n)) * 512;
        v8f acc[8];
#pragma unroll
        for (int h = 0; h < 8; ++h) {
            V16U a;
            a.h[0] = *(const v8bf*)(q + qbase + h * 32 + lh * 16);
            a.h[1] = *(const v8bf*)(q + qbase + h * 32 + 8 + lh * 16);
            const v16bf b = *(const v16bf*)(kv + kbase + h * 32 + lh * 16);
            acc[h] = __builtin_amdgcn_wmma_f32_16x16x32_bf16(
                false, a.v, false, b, (short)0, vz8(), false, false);
        }
#pragma unroll
        for (int r = 0; r < 8; ++r) {
            float m = acc[0][r];
#pragma unroll
            for (int h = 1; h < 8; ++h) m = fmaxf(m, acc[h][r]);
            float e8[8], sum = 0.0f;
#pragma unroll
            for (int h = 0; h < 8; ++h) { e8[h] = __expf(acc[h][r] - m); sum += e8[h]; }
            const float inv = 1.0f / sum;
            const int ii = ifr * 16 + r + lh * 8;
            const int jj = jfr * 16 + l15;
#pragma unroll
            for (int h = 0; h < 8; ++h)
                s_s[h * 4096 + ii * 64 + jj] = (__bf16)(e8[h] * inv);
        }
    }
    __syncthreads();

    // ---- out = S @ v per head; wave -> (i-frag, c-frag) ----
    const int ifr = wave & 3;
    const int cf  = wave >> 2;   // 0/1 -> c halves of CH=32
#pragma unroll
    for (int h = 0; h < 8; ++h) {
        v8f acc = vz8();
#pragma unroll
        for (int ks = 0; ks < 64; ks += 32) {
            V16U a;
            const __bf16* sp = s_s + h * 4096 + (ifr * 16 + l15) * 64 + ks;
            a.h[0] = *(const v8bf*)(sp + lh * 16);
            a.h[1] = *(const v8bf*)(sp + 8 + lh * 16);
            const v16bf b = *(const v16bf*)(s_vt + h * 2048 + (cf * 16 + l15) * 64 +
                                            ks + lh * 16);
            acc = __builtin_amdgcn_wmma_f32_16x16x32_bf16(
                false, a.v, false, b, (short)0, acc, false, false);
        }
#pragma unroll
        for (int r = 0; r < 8; ++r) {
            const int irow = itile * 64 + ifr * 16 + r + lh * 8;
            const int col  = h * 32 + cf * 16 + l15;
            const size_t off = ((size_t)irow * NSEQ + n) * CDIM + col;
            const float g = (float)gate[off];
            const float sig = 1.0f / (1.0f + __expf(-g));
            att[off] = (__bf16)(acc[r] * sig);
        }
    }
}

// ---------------------------------------------------------------------------
// Host orchestration
// ---------------------------------------------------------------------------
extern "C" void kernel_launch(void* const* d_in, const int* in_sizes, int n_in,
                              void* d_out, int out_size, void* d_ws, size_t ws_size,
                              hipStream_t stream) {
    const float* embed  = (const float*)d_in[0];
    const float* memry  = (const float*)d_in[1];
    const int*   rmask  = (const int*)d_in[2];
    // d_in[3] memory_mask: additive mask is constant across the softmax axis
    // (h) in the reference, so it cancels exactly -> unused.
    const float* lnq_s  = (const float*)d_in[4];
    const float* lnq_b  = (const float*)d_in[5];
    const float* lnkv_s = (const float*)d_in[6];
    const float* lnkv_b = (const float*)d_in[7];
    const float* Wq     = (const float*)d_in[8];
    const float* Wkv    = (const float*)d_in[9];
    const float* Wgate  = (const float*)d_in[10];
    const float* Wout   = (const float*)d_in[11];
    const float* lnf_s  = (const float*)d_in[12];
    const float* lnf_b  = (const float*)d_in[13];
    const float* W1     = (const float*)d_in[14];
    const float* b1     = (const float*)d_in[15];
    const float* W2     = (const float*)d_in[16];
    const float* b2     = (const float*)d_in[17];
    float* out = (float*)d_out;

    char* ws = (char*)d_ws;
    size_t off = 0;
    auto alloc = [&](size_t bytes) {
        void* p = ws + off;
        off += (bytes + 255) & ~(size_t)255;
        return p;
    };

    __bf16* qe   = (__bf16*)alloc((size_t)ROWS_Q * CDIM * 2);   // reused as att
    __bf16* km   = (__bf16*)alloc((size_t)ROWS_KV * CDIM * 2);
    __bf16* qb   = (__bf16*)alloc((size_t)ROWS_Q * CDIM * 2);   // reused as xln
    __bf16* gb   = (__bf16*)alloc((size_t)ROWS_Q * CDIM * 2);
    __bf16* kvb  = (__bf16*)alloc((size_t)ROWS_KV * 512 * 2);
    float*  xf   = (float*)alloc((size_t)ROWS_Q * CDIM * 4);
    __bf16* h1b  = (__bf16*)alloc((size_t)ROWS_Q * 512 * 2);
    __bf16* WqT   = (__bf16*)alloc(256 * 256 * 2);
    __bf16* WkvT  = (__bf16*)alloc(256 * 512 * 2);
    __bf16* WgT   = (__bf16*)alloc(256 * 256 * 2);
    __bf16* WoT   = (__bf16*)alloc(256 * 256 * 2);
    __bf16* W1T   = (__bf16*)alloc(256 * 512 * 2);
    __bf16* W2T   = (__bf16*)alloc(512 * 256 * 2);
    __bf16* attb = qe;   // lifetime-disjoint reuse
    __bf16* xln  = qb;

    // weight transposes (bf16)
    wtrans_kernel<<<256, 256, 0, stream>>>(Wq,   WqT,  256, 256);
    wtrans_kernel<<<512, 256, 0, stream>>>(Wkv,  WkvT, 256, 512);
    wtrans_kernel<<<256, 256, 0, stream>>>(Wgate,WgT,  256, 256);
    wtrans_kernel<<<256, 256, 0, stream>>>(Wout, WoT,  256, 256);
    wtrans_kernel<<<512, 256, 0, stream>>>(W1,   W1T,  256, 512);
    wtrans_kernel<<<512, 256, 0, stream>>>(W2,   W2T,  512, 256);

    // LayerNorms -> bf16
    ln256_kernel<<<ROWS_Q / 8, 256, 0, stream>>>(embed, lnq_s, lnq_b, qe, ROWS_Q);
    ln256_kernel<<<ROWS_KV / 8, 256, 0, stream>>>(memry, lnkv_s, lnkv_b, km, ROWS_KV);

    // projections
    dim3 gq(ROWS_Q / 64, 256 / 64);
    dim3 gkv(ROWS_KV / 64, 512 / 64);
    gemm_bf16_kernel<false, false, 0><<<gq, 128, 0, stream>>>(
        qe, WqT, nullptr, nullptr, nullptr, nullptr, qb, ROWS_Q, 256, 256);
    gemm_bf16_kernel<false, false, 0><<<gq, 128, 0, stream>>>(
        qe, WgT, nullptr, nullptr, nullptr, nullptr, gb, ROWS_Q, 256, 256);
    gemm_bf16_kernel<false, false, 0><<<gkv, 128, 0, stream>>>(
        km, WkvT, nullptr, nullptr, nullptr, nullptr, kvb, ROWS_KV, 512, 256);

    // fused attention + softmax(h) + gating
    attn_kernel<<<dim3(4, NSEQ), 256, 98304, stream>>>(qb, kvb, gb, attb);

    // Wout + masked residual -> x (f32)
    gemm_bf16_kernel<false, false, 1><<<gq, 128, 0, stream>>>(
        attb, WoT, nullptr, embed, rmask, xf, nullptr, ROWS_Q, 256, 256);

    // FFN
    ln256_kernel<<<ROWS_Q / 8, 256, 0, stream>>>(xf, lnf_s, lnf_b, xln, ROWS_Q);
    dim3 g1(ROWS_Q / 64, 512 / 64);
    gemm_bf16_kernel<true, true, 0><<<g1, 128, 0, stream>>>(
        xln, W1T, b1, nullptr, nullptr, nullptr, h1b, ROWS_Q, 512, 256);
    gemm_bf16_kernel<false, true, 1><<<gq, 128, 0, stream>>>(
        h1b, W2T, b2, xf, rmask, out, nullptr, ROWS_Q, 256, 512);
}
